// Encoder_50757923504656
// MI455X (gfx1250) — matrix operand order repaired and checked by reading, not compiled, not run on hardware
//
#include <hip/hip_runtime.h>
#include <hip/hip_bf16.h>
#include <math.h>

#define Bdim 2
#define Sdim 1024
#define Ddim 1024
#define Hdim 16
#define HDdim 64
#define FFdim 4096
#define Ldim 4

typedef __attribute__((ext_vector_type(16))) __bf16 v16bf;
typedef __attribute__((ext_vector_type(8)))  float  v8f;

union FragA { unsigned int u[8]; v16bf v; };

__device__ __forceinline__ unsigned short f2bf(float f) {
  unsigned int u = __float_as_uint(f);
  unsigned int r = (u + 0x7FFFu + ((u >> 16) & 1u)) >> 16;   // RNE truncate
  return (unsigned short)r;
}
__device__ __forceinline__ unsigned int packbf(float a, float b) {
  return (unsigned int)f2bf(a) | ((unsigned int)f2bf(b) << 16);
}
__device__ __forceinline__ v8f wmma_bf16(const FragA& a, const FragA& b, v8f c) {
  return __builtin_amdgcn_wmma_f32_16x16x32_bf16(false, a.v, false, b.v,
                                                 (short)0, c, false, false);
}
// low 32 bits of a flat shared pointer = LDS byte offset for this workgroup
__device__ __forceinline__ unsigned lds_off(const void* p) {
  return (unsigned)(size_t)p;
}

// ---------------------------------------------------------------------------
// bf16 GEMM with async-LDS double-buffered staging:
//   C[MxN] = act(Abf[MxK] @ Bt[NxK]^T + bias) + res
// Abf row-major bf16, Bt = B transposed (row n holds K contiguous bf16).
// Block tile 128x128, K-step 32, 8 waves (2x4), wave tile 64x32.
// ---------------------------------------------------------------------------
#define BM 128
#define BN 128
#define BK 32
#define LSTR 40                 // LDS row stride (ushort): 80B, 16B-aligned
#define TILE_USH (BM * LSTR)    // ushorts per LDS tile buffer

__global__ __launch_bounds__(256)
void gemm_bf_kernel(const unsigned short* __restrict__ Abf,
                    const unsigned short* __restrict__ Bt,
                    const float* __restrict__ bias, const float* __restrict__ res,
                    float* __restrict__ C, int M, int N, int K, int relu)
{
  __shared__ unsigned short As[2 * TILE_USH];
  __shared__ unsigned short Bs[2 * TILE_USH];

  const int tid  = threadIdx.x;
  const int lane = tid & 31;
  const int wave = tid >> 5;
  const int hlf  = lane >> 4;
  const int l16  = lane & 15;

  const int m0 = blockIdx.y * BM;
  const int n0 = blockIdx.x * BN;
  const int wm = wave >> 2;   // 0..1
  const int wn = wave & 3;    // 0..3

  v8f acc[4][2];
  #pragma unroll
  for (int i = 0; i < 4; i++)
    #pragma unroll
    for (int j = 0; j < 2; j++)
      #pragma unroll
      for (int e = 0; e < 8; e++) acc[i][j][e] = 0.f;

  // staging: each thread DMAs two 16B chunks of A and two of B per K-step.
  // row = tid>>1 (0..127), chunk base sc in {0,2} -> elems {0,16}
  const int srow = tid >> 1;
  const int sc8  = (tid & 1) * 16;   // element offset of first chunk
  const unsigned lA0 = lds_off(&As[srow * LSTR + sc8]);
  const unsigned lB0 = lds_off(&Bs[srow * LSTR + sc8]);
  const unsigned short* gA = Abf + (size_t)(m0 + srow) * K + sc8;
  const unsigned short* gB = Bt  + (size_t)(n0 + srow) * K + sc8;

  auto issue = [&](int buf, int kb) {
    const unsigned la = lA0 + (unsigned)(buf * (TILE_USH * 2)); // bytes
    const unsigned lb = lB0 + (unsigned)(buf * (TILE_USH * 2));
    const unsigned short* pa = gA + kb;
    const unsigned short* pb = gB + kb;
    asm volatile("global_load_async_to_lds_b128 %0, %1, off"
                 :: "v"(la), "v"(pa) : "memory");
    asm volatile("global_load_async_to_lds_b128 %0, %1, off"
                 :: "v"(la + 16u), "v"(pa + 8) : "memory");
    asm volatile("global_load_async_to_lds_b128 %0, %1, off"
                 :: "v"(lb), "v"(pb) : "memory");
    asm volatile("global_load_async_to_lds_b128 %0, %1, off"
                 :: "v"(lb + 16u), "v"(pb + 8) : "memory");
  };

  issue(0, 0);
  int buf = 0;
  for (int kb = 0; kb < K; kb += BK, buf ^= 1) {
    const bool more = (kb + BK) < K;
    if (more) {
      issue(buf ^ 1, kb + BK);                       // prefetch next tile
      asm volatile("s_wait_asynccnt 0x4" ::: "memory"); // current tile landed
    } else {
      asm volatile("s_wait_asynccnt 0x0" ::: "memory");
    }
    __syncthreads();

    const unsigned short* Ab = &As[buf * TILE_USH];
    const unsigned short* Bb = &Bs[buf * TILE_USH];
    FragA af[4], bfr[2];
    #pragma unroll
    for (int mt = 0; mt < 4; mt++) {
      const int rb = wm * 64 + mt * 16 + l16;
      #pragma unroll
      for (int j = 0; j < 8; j++) {
        const int kk = (j < 4 ? 0 : 16) + hlf * 8 + (j & 3) * 2;  // A 16x32 layout
        af[mt].u[j] = *(const unsigned int*)&Ab[rb * LSTR + kk];
      }
    }
    #pragma unroll
    for (int nt = 0; nt < 2; nt++) {
      const int cb = wn * 32 + nt * 16 + l16;
      #pragma unroll
      for (int j = 0; j < 8; j++) {
        const int kk = (j < 4 ? 0 : 16) + hlf * 8 + (j & 3) * 2;  // B 32x16 layout
        bfr[nt].u[j] = *(const unsigned int*)&Bb[cb * LSTR + kk];
      }
    }
    #pragma unroll
    for (int mt = 0; mt < 4; mt++)
      #pragma unroll
      for (int nt = 0; nt < 2; nt++)
        acc[mt][nt] = wmma_bf16(af[mt], bfr[nt], acc[mt][nt]);
    __syncthreads();
  }

  // epilogue: bias, relu, residual
  #pragma unroll
  for (int mt = 0; mt < 4; mt++) {
    #pragma unroll
    for (int nt = 0; nt < 2; nt++) {
      const int col = n0 + wn * 32 + nt * 16 + l16;
      const float bv = bias ? bias[col] : 0.f;
      #pragma unroll
      for (int e = 0; e < 8; e++) {
        const int row = m0 + wm * 64 + mt * 16 + e + hlf * 8;
        float vo = acc[mt][nt][e] + bv;
        if (relu) vo = fmaxf(vo, 0.f);
        if (res)  vo += res[(size_t)row * N + col];
        C[(size_t)row * N + col] = vo;
      }
    }
  }
  (void)M;
}

// ---------------------------------------------------------------------------
// f32 -> bf16 elementwise convert (row-major), 4 elems/thread
// ---------------------------------------------------------------------------
__global__ __launch_bounds__(256)
void cvt_kernel(const float* __restrict__ in, unsigned short* __restrict__ out,
                int n)
{
  const int i = (blockIdx.x * 256 + threadIdx.x) * 4;
  if (i < n) {
    const float4 v = *(const float4*)&in[i];
    *(unsigned int*)&out[i]     = packbf(v.x, v.y);
    *(unsigned int*)&out[i + 2] = packbf(v.z, v.w);
  }
}

// ---------------------------------------------------------------------------
// Transpose + convert: W[K][N] f32 -> Wt[N][K] bf16 (32x32 LDS tiles)
// ---------------------------------------------------------------------------
__global__ __launch_bounds__(256)
void tcvt_kernel(const float* __restrict__ W, unsigned short* __restrict__ Wt,
                 int K, int N)
{
  __shared__ float Ts[32][33];
  const int tid = threadIdx.x;
  const int n0 = blockIdx.x * 32;
  const int k0 = blockIdx.y * 32;
  const int c  = tid & 31;
  const int r4 = tid >> 5;
  #pragma unroll
  for (int p = 0; p < 4; p++) {
    const int r = p * 8 + r4;
    Ts[r][c] = W[(size_t)(k0 + r) * N + n0 + c];
  }
  __syncthreads();
  const int kx = (tid & 15) * 2;
  const int ny = tid >> 4;
  #pragma unroll
  for (int p = 0; p < 2; p++) {
    const int n = p * 16 + ny;
    *(unsigned int*)&Wt[(size_t)(n0 + n) * K + k0 + kx] =
        packbf(Ts[kx][n], Ts[kx + 1][n]);
  }
}

// ---------------------------------------------------------------------------
// Flash attention: block = (128 q-rows, head, batch); 8 waves x 16 q-rows.
// ---------------------------------------------------------------------------
__global__ __launch_bounds__(256)
void attn_kernel(const float* __restrict__ qkv, const float* __restrict__ mask,
                 float* __restrict__ out)
{
  __shared__ unsigned short Ks[32 * 72];       // [kcol][d]  bf16
  __shared__ unsigned short Vs[64 * 40];       // [d][kcol]  bf16 (transposed)
  __shared__ unsigned short Ps[8 * 16 * 40];   // per-wave P tile 16x32 bf16

  const int tid  = threadIdx.x;
  const int lane = tid & 31;
  const int wave = tid >> 5;
  const int hlf  = lane >> 4;
  const int l16  = lane & 15;

  const int b  = blockIdx.z;
  const int h  = blockIdx.y;
  const int q0 = blockIdx.x * 128 + wave * 16;

  const size_t rstride = 3 * Ddim;
  const size_t hq = (size_t)h * 3 * HDdim;

  FragA qa[2];
  {
    const float* qptr = qkv + ((size_t)(b * Sdim + q0 + l16)) * rstride + hq;
    #pragma unroll
    for (int s = 0; s < 2; s++)
      #pragma unroll
      for (int j = 0; j < 8; j++) {
        const int kk = s * 32 + (j < 4 ? 0 : 16) + hlf * 8 + (j & 3) * 2;
        const float2 v = *(const float2*)&qptr[kk];
        qa[s].u[j] = packbf(v.x, v.y);
      }
  }

  const float scale = 0.125f;
  v8f o[4];
  #pragma unroll
  for (int nt = 0; nt < 4; nt++)
    #pragma unroll
    for (int e = 0; e < 8; e++) o[nt][e] = 0.f;
  float mrun[8], lrun[8];
  #pragma unroll
  for (int e = 0; e < 8; e++) { mrun[e] = -INFINITY; lrun[e] = 0.f; }

  unsigned short* Psw = &Ps[wave * 16 * 40];
  const int str = tid >> 4;
  const int stc = (tid & 15) * 4;

  for (int kt = 0; kt < Sdim / 32; kt++) {
    __syncthreads();
    #pragma unroll
    for (int p = 0; p < 2; p++) {
      const int kc = p * 16 + str;
      const float* kvp = qkv + ((size_t)(b * Sdim + kt * 32 + kc)) * rstride + hq;
      const float4 kv = *(const float4*)&kvp[64 + stc];
      *(unsigned int*)&Ks[kc * 72 + stc]     = packbf(kv.x, kv.y);
      *(unsigned int*)&Ks[kc * 72 + stc + 2] = packbf(kv.z, kv.w);
      const float4 vv = *(const float4*)&kvp[128 + stc];
      Vs[(stc + 0) * 40 + kc] = f2bf(vv.x);
      Vs[(stc + 1) * 40 + kc] = f2bf(vv.y);
      Vs[(stc + 2) * 40 + kc] = f2bf(vv.z);
      Vs[(stc + 3) * 40 + kc] = f2bf(vv.w);
    }
    __syncthreads();

    v8f s0, s1;
    #pragma unroll
    for (int e = 0; e < 8; e++) { s0[e] = 0.f; s1[e] = 0.f; }
    #pragma unroll
    for (int s = 0; s < 2; s++) {
      FragA bk0, bk1;
      #pragma unroll
      for (int j = 0; j < 8; j++) {
        const int kk = s * 32 + (j < 4 ? 0 : 16) + hlf * 8 + (j & 3) * 2;
        bk0.u[j] = *(const unsigned int*)&Ks[l16 * 72 + kk];
        bk1.u[j] = *(const unsigned int*)&Ks[(16 + l16) * 72 + kk];
      }
      s0 = wmma_bf16(qa[s], bk0, s0);
      s1 = wmma_bf16(qa[s], bk1, s1);
    }

    float p0[8], p1[8];
    #pragma unroll
    for (int e = 0; e < 8; e++) {
      const int qrow = q0 + e + 8 * hlf;
      const int kc0  = kt * 32 + l16;
      float a0 = s0[e] * scale + mask[(size_t)qrow * Sdim + kc0];
      float a1 = s1[e] * scale + mask[(size_t)qrow * Sdim + kc0 + 16];
      float mx = fmaxf(a0, a1);
      #pragma unroll
      for (int off = 8; off >= 1; off >>= 1)
        mx = fmaxf(mx, __shfl_xor(mx, off, 32));
      const float mnew = fmaxf(mrun[e], mx);
      const float fac  = __expf(mrun[e] - mnew);
      const float e0   = __expf(a0 - mnew);
      const float e1   = __expf(a1 - mnew);
      float ls = e0 + e1;
      #pragma unroll
      for (int off = 8; off >= 1; off >>= 1)
        ls += __shfl_xor(ls, off, 32);
      lrun[e] = lrun[e] * fac + ls;
      mrun[e] = mnew;
      p0[e] = e0; p1[e] = e1;
      #pragma unroll
      for (int nt = 0; nt < 4; nt++) o[nt][e] *= fac;
    }
    #pragma unroll
    for (int e = 0; e < 8; e++) {
      const int r = e + 8 * hlf;
      Psw[r * 40 + l16]      = f2bf(p0[e]);
      Psw[r * 40 + 16 + l16] = f2bf(p1[e]);
    }
    __syncthreads();

    FragA pa;
    #pragma unroll
    for (int j = 0; j < 8; j++) {
      const int kk = (j < 4 ? 0 : 16) + hlf * 8 + (j & 3) * 2;
      pa.u[j] = *(const unsigned int*)&Psw[l16 * 40 + kk];
    }
    #pragma unroll
    for (int nt = 0; nt < 4; nt++) {
      FragA bv;
      #pragma unroll
      for (int j = 0; j < 8; j++) {
        const int kk = (j < 4 ? 0 : 16) + hlf * 8 + (j & 3) * 2;
        bv.u[j] = *(const unsigned int*)&Vs[(nt * 16 + l16) * 40 + kk];
      }
      o[nt] = wmma_bf16(pa, bv, o[nt]);
    }
  }

  #pragma unroll
  for (int e = 0; e < 8; e++) {
    const float inv = 1.0f / lrun[e];
    const int qrow  = q0 + e + 8 * hlf;
    float* op = out + ((size_t)(b * Sdim + qrow)) * Ddim + h * HDdim;
    #pragma unroll
    for (int nt = 0; nt < 4; nt++)
      op[nt * 16 + l16] = o[nt][e] * inv;
  }
}

// ---------------------------------------------------------------------------
// Row LayerNorm over D=1024: one 256-thread block per row, 4 elems/thread.
// ---------------------------------------------------------------------------
__global__ __launch_bounds__(256)
void ln_kernel(const float* __restrict__ x, const float* __restrict__ g,
               const float* __restrict__ bb, float* __restrict__ y)
{
  __shared__ float pm[8], pv[8];
  const int tid = threadIdx.x;
  const int row = blockIdx.x;
  const int c   = tid * 4;
  const float4 v = *(const float4*)&x[(size_t)row * Ddim + c];
  float s = v.x + v.y + v.z + v.w;
  #pragma unroll
  for (int off = 16; off >= 1; off >>= 1) s += __shfl_xor(s, off, 32);
  if ((tid & 31) == 0) pm[tid >> 5] = s;
  __syncthreads();
  float mean = 0.f;
  #pragma unroll
  for (int i = 0; i < 8; i++) mean += pm[i];
  mean *= (1.0f / Ddim);
  const float d0 = v.x - mean, d1 = v.y - mean, d2 = v.z - mean, d3 = v.w - mean;
  float sq = d0 * d0 + d1 * d1 + d2 * d2 + d3 * d3;
  #pragma unroll
  for (int off = 16; off >= 1; off >>= 1) sq += __shfl_xor(sq, off, 32);
  if ((tid & 31) == 0) pv[tid >> 5] = sq;
  __syncthreads();
  float var = 0.f;
  #pragma unroll
  for (int i = 0; i < 8; i++) var += pv[i];
  var *= (1.0f / Ddim);
  const float inv = rsqrtf(var + 1e-5f);
  const float4 gv = *(const float4*)&g[c];
  const float4 bv = *(const float4*)&bb[c];
  float4 r;
  r.x = gv.x * d0 * inv + bv.x;
  r.y = gv.y * d1 * inv + bv.y;
  r.z = gv.z * d2 * inv + bv.z;
  r.w = gv.w * d3 * inv + bv.w;
  *(float4*)&y[(size_t)row * Ddim + c] = r;
}

// ---------------------------------------------------------------------------
extern "C" void kernel_launch(void* const* d_in, const int* in_sizes, int n_in,
                              void* d_out, int out_size, void* d_ws, size_t ws_size,
                              hipStream_t stream)
{
  const float* x    = (const float*)d_in[0];
  const float* mask = (const float*)d_in[1];
  const float* Wqkv = (const float*)d_in[2];
  const float* bqkv = (const float*)d_in[3];
  const float* Wo   = (const float*)d_in[4];
  const float* bo   = (const float*)d_in[5];
  const float* g1   = (const float*)d_in[6];
  const float* be1  = (const float*)d_in[7];
  const float* W1   = (const float*)d_in[8];
  const float* bf1  = (const float*)d_in[9];
  const float* W2   = (const float*)d_in[10];
  const float* bf2  = (const float*)d_in[11];
  const float* g2   = (const float*)d_in[12];
  const float* be2  = (const float*)d_in[13];
  float* outp = (float*)d_out;

  const size_t Mrows = (size_t)Bdim * Sdim;   // 2048
  float* ws   = (float*)d_ws;
  float* qkv  = ws;                           // 2048 x 3072 f32
  float* attn = qkv  + Mrows * 3 * Ddim;      // 2048 x 1024 f32
  float* tmp  = attn + Mrows * Ddim;
  float* ln1  = tmp  + Mrows * Ddim;
  float* ff   = ln1  + Mrows * Ddim;          // 2048 x 4096 f32
  float* hbuf = ff   + Mrows * FFdim;         // 2048 x 1024 f32
  unsigned short* wq_t = (unsigned short*)(hbuf + Mrows * Ddim); // [3072][1024]
  unsigned short* wo_t = wq_t + (size_t)3 * Ddim * Ddim;         // [1024][1024]
  unsigned short* w1_t = wo_t + (size_t)Ddim * Ddim;             // [4096][1024]
  unsigned short* w2_t = w1_t + (size_t)FFdim * Ddim;            // [1024][4096]
  unsigned short* abuf = w2_t + (size_t)Ddim * FFdim;            // 2048 x 4096 bf16

  const dim3 blk(256);
  const int nD  = (int)(Mrows * Ddim);        // 2,097,152
  const int nFF = (int)(Mrows * FFdim);       // 8,388,608

  for (int l = 0; l < Ldim; l++) {
    const float* hin  = (l == 0) ? x : hbuf;
    float*       hout = (l == Ldim - 1) ? outp : hbuf;

    // per-layer weight transpose+convert into L2-resident bf16 scratch
    tcvt_kernel<<<dim3(3 * Ddim / 32, Ddim / 32), blk, 0, stream>>>(
        Wqkv + (size_t)l * Ddim * 3 * Ddim, wq_t, Ddim, 3 * Ddim);
    tcvt_kernel<<<dim3(Ddim / 32, Ddim / 32), blk, 0, stream>>>(
        Wo + (size_t)l * Ddim * Ddim, wo_t, Ddim, Ddim);
    tcvt_kernel<<<dim3(FFdim / 32, Ddim / 32), blk, 0, stream>>>(
        W1 + (size_t)l * Ddim * FFdim, w1_t, Ddim, FFdim);
    tcvt_kernel<<<dim3(Ddim / 32, FFdim / 32), blk, 0, stream>>>(
        W2 + (size_t)l * FFdim * Ddim, w2_t, FFdim, Ddim);

    // qkv = hin @ Wqkv + bqkv
    cvt_kernel<<<dim3(nD / 1024), blk, 0, stream>>>(hin, abuf, nD);
    gemm_bf_kernel<<<dim3(3 * Ddim / BN, (unsigned)(Mrows / BM)), blk, 0, stream>>>(
        abuf, wq_t, bqkv + (size_t)l * 3 * Ddim, nullptr, qkv,
        (int)Mrows, 3 * Ddim, Ddim, 0);

    // attn = softmax(Q K^T / sqrt(hd) + mask) V
    attn_kernel<<<dim3(Sdim / 128, Hdim, Bdim), blk, 0, stream>>>(qkv, mask, attn);

    // tmp = attn @ Wo + bo + hin
    cvt_kernel<<<dim3(nD / 1024), blk, 0, stream>>>(attn, abuf, nD);
    gemm_bf_kernel<<<dim3(Ddim / BN, (unsigned)(Mrows / BM)), blk, 0, stream>>>(
        abuf, wo_t, bo + (size_t)l * Ddim, hin, tmp, (int)Mrows, Ddim, Ddim, 0);

    // ln1 = LN(tmp)
    ln_kernel<<<dim3((unsigned)Mrows), blk, 0, stream>>>(
        tmp, g1 + (size_t)l * Ddim, be1 + (size_t)l * Ddim, ln1);

    // ff = relu(ln1 @ W1 + bf1)
    cvt_kernel<<<dim3(nD / 1024), blk, 0, stream>>>(ln1, abuf, nD);
    gemm_bf_kernel<<<dim3(FFdim / BN, (unsigned)(Mrows / BM)), blk, 0, stream>>>(
        abuf, w1_t, bf1 + (size_t)l * FFdim, nullptr, ff,
        (int)Mrows, FFdim, Ddim, 1);

    // tmp = ff @ W2 + bf2 + ln1
    cvt_kernel<<<dim3(nFF / 1024), blk, 0, stream>>>(ff, abuf, nFF);
    gemm_bf_kernel<<<dim3(Ddim / BN, (unsigned)(Mrows / BM)), blk, 0, stream>>>(
        abuf, w2_t, bf2 + (size_t)l * Ddim, ln1, tmp, (int)Mrows, Ddim, FFdim, 0);

    // hout = LN(tmp)
    ln_kernel<<<dim3((unsigned)Mrows), blk, 0, stream>>>(
        tmp, g2 + (size_t)l * Ddim, be2 + (size_t)l * Ddim, hout);
  }
  (void)in_sizes; (void)n_in; (void)out_size; (void)ws_size;
}
